// LSTMEncDec_46746424049706
// MI455X (gfx1250) — compile-verified, matrix operand order checked by reading, not run on hardware
//
#include <hip/hip_runtime.h>
#include <hip/hip_bf16.h>

typedef __attribute__((ext_vector_type(16))) __bf16 v16bf;
typedef __attribute__((ext_vector_type(8)))  float  v8f;

union Frag { v16bf v; uint4 q[2]; };

#define WMMA_BF16(a, b, c) \
  __builtin_amdgcn_wmma_f32_16x16x32_bf16(false, (a), false, (b), (short)0, (c), false, false)

// Async global -> LDS 16-byte copy (CDNA5 GLOBAL_LOAD_ASYNC_TO_LDS_B128,
// tracked by ASYNCcnt). LDS operand is the addrspace(3) byte offset, i.e. the
// low 32 bits of the generic pointer to a __shared__ object.
__device__ __forceinline__ void async_copy16(const __bf16* gp, __bf16* lp) {
  unsigned lds = (unsigned)(unsigned long long)lp;
  unsigned long long ga = (unsigned long long)gp;
  asm volatile("global_load_async_to_lds_b128 %0, %1, off"
               :: "v"(lds), "v"(ga) : "memory");
}
#define WAIT_ASYNC0() asm volatile("s_wait_asynccnt 0x0" ::: "memory")
#define WAIT_ASYNC1() asm volatile("s_wait_asynccnt 0x1" ::: "memory")

__device__ __forceinline__ float sigf(float x) { return 1.0f / (1.0f + __expf(-x)); }

// ---------------------------------------------------------------- utilities
__global__ void k_zero_bf16(__bf16* p, int n) {
  int i = blockIdx.x * blockDim.x + threadIdx.x;
  if (i < n) p[i] = (__bf16)0.0f;
}
__global__ void k_zero_f32(float* p, int n) {
  int i = blockIdx.x * blockDim.x + threadIdx.x;
  if (i < n) p[i] = 0.0f;
}

// Cast fp32 weight (N x K, row major) -> bf16 (Npad x Kpad), zero padded.
__global__ void k_cast_pad(const float* __restrict__ w, __bf16* __restrict__ o,
                           int N, int K, int Npad, int Kpad) {
  long i = (long)blockIdx.x * blockDim.x + threadIdx.x;
  long tot = (long)Npad * Kpad;
  if (i >= tot) return;
  int n = (int)(i / Kpad), k = (int)(i % Kpad);
  float v = (n < N && k < K) ? w[(long)n * K + k] : 0.0f;
  o[i] = (__bf16)v;
}

// Embedding gather + cast: out[r][e] = emb[idx[r]][e], rows >= rows_valid zeroed.
__global__ void k_gather_cast(const int* __restrict__ idx, const float* __restrict__ emb,
                              __bf16* __restrict__ o, int rows_valid, int Mpad) {
  long i = (long)blockIdx.x * blockDim.x + threadIdx.x;
  long tot = (long)Mpad * 256;
  if (i >= tot) return;
  int r = (int)(i >> 8), e = (int)(i & 255);
  float v = 0.0f;
  if (r < rows_valid) v = emb[(long)idx[r] * 256 + e];
  o[i] = (__bf16)v;
}

// ------------------------------------------------- generic bf16 WMMA GEMM
// C[M x Nld] (fp32) = A[Mpad x Kpad] * Bt[Npad x Kpad]^T + bias.
// Block: 256 threads (8 waves); block tile 128(M) x 64(N).
// B tile double-buffered in LDS, filled by async global->LDS DMA that
// overlaps the WMMA work on the previous tile.
__global__ __launch_bounds__(256) void k_gemm_bias(
    const __bf16* __restrict__ A, const __bf16* __restrict__ Bt,
    const float* __restrict__ bias, float* __restrict__ C,
    int M, int N, int Nld, int Kpad) {
  __shared__ __align__(16) __bf16 btile[2][64 * 32];
  const int tid = threadIdx.x;
  const int w = tid >> 5, lane = tid & 31;
  const int nrow = lane & 15, hi = lane >> 4;
  const int m0 = blockIdx.y * 128 + w * 16;
  const int n0 = blockIdx.x * 64;
  const int nl = tid >> 2;          // 0..63: B row staged by this thread
  const int kc = (tid & 3) * 8;     // 8-element chunk within 32-wide k slab

  v8f acc[4];
  for (int nt = 0; nt < 4; ++nt)
    for (int v = 0; v < 8; ++v) acc[nt][v] = 0.0f;

  const int ksteps = Kpad >> 5;
  const __bf16* brow = Bt + (long)(n0 + nl) * Kpad + kc;
  async_copy16(brow, &btile[0][nl * 32 + kc]);   // stage tile 0

  for (int kk = 0; kk < ksteps; ++kk) {
    const int cur = kk & 1;
    if (kk + 1 < ksteps) {
      async_copy16(brow + (kk + 1) * 32, &btile[cur ^ 1][nl * 32 + kc]);
      WAIT_ASYNC1();                 // own transfer for tile kk complete
    } else {
      WAIT_ASYNC0();
    }
    __syncthreads();                 // tile kk visible to all waves

    Frag a;
    const int kb = kk * 32 + hi * 8;
    const __bf16* ap = A + (long)(m0 + nrow) * Kpad + kb;
    a.q[0] = *(const uint4*)(ap);
    a.q[1] = *(const uint4*)(ap + 16);
    const __bf16* bt = &btile[cur][0];
    for (int nt = 0; nt < 4; ++nt) {
      Frag b;
      const __bf16* bp = bt + (nt * 16 + nrow) * 32 + hi * 8;
      b.q[0] = *(const uint4*)(bp);
      b.q[1] = *(const uint4*)(bp + 16);
      acc[nt] = WMMA_BF16(a.v, b.v, acc[nt]);
    }
    __syncthreads();                 // done reading before tile kk+2 overwrite
  }
  for (int nt = 0; nt < 4; ++nt)
    for (int v = 0; v < 8; ++v) {
      int m = m0 + v + 8 * hi;
      int n = n0 + nt * 16 + nrow;
      if (m < M && n < N) C[(long)m * Nld + n] = acc[nt][v] + bias[n];
    }
}

// ---------------------------------------------------- persistent LSTM scan
// One workgroup (32 waves). h double-buffered in LDS as bf16 [32][544];
// c lives in registers. Per step: g = xg[t] + h @ whh^T, gates -> h,c.
// xg: [S*32][2064] fp32. whh: [2112][544] bf16. hs: [S*32][544] bf16 out.
__global__ __launch_bounds__(1024) void k_lstm_rec(
    const float* __restrict__ xg, const __bf16* __restrict__ whh,
    const float* __restrict__ h0, const float* __restrict__ c0,
    __bf16* __restrict__ hs, float* __restrict__ hN, float* __restrict__ cN,
    int S) {
  __shared__ __align__(16) __bf16 hbuf[2][32 * 544];
  const int tid = threadIdx.x;
  const int w = tid >> 5, lane = tid & 31;
  const int nrow = lane & 15, hi = lane >> 4;

  for (int i = tid; i < 32 * 544; i += 1024) {
    int r = i / 544, cc = i % 544;
    float v = (h0 != nullptr && cc < 516) ? h0[r * 516 + cc] : 0.0f;
    hbuf[0][i] = (__bf16)v;
    hbuf[1][i] = (__bf16)0.0f;
  }

  float creg[2][2][8];
  for (int slot = 0; slot < 2; ++slot) {
    int tile = w + slot * 32;
    for (int mt = 0; mt < 2; ++mt)
      for (int v = 0; v < 8; ++v) {
        float cv = 0.0f;
        if (c0 != nullptr && tile < 33) {
          int m = mt * 16 + v + 8 * hi;
          int tc = tile * 16 + nrow;
          if (tc < 516) cv = c0[m * 516 + tc];
        }
        creg[slot][mt][v] = cv;
      }
  }
  __syncthreads();

  for (int t = 0; t < S; ++t) {
    const int cur = t & 1, nxt = cur ^ 1;
    const float* xgt = xg + (long)t * 32 * 2064;
    for (int tile = w; tile < 33; tile += 32) {
      const int slot = tile >> 5;
      const int tc = tile * 16 + nrow;
      const bool cok = (tc < 516);
      v8f acc[4][2];
      for (int g = 0; g < 4; ++g)
        for (int mt = 0; mt < 2; ++mt)
          for (int v = 0; v < 8; ++v) {
            int m = mt * 16 + v + 8 * hi;
            acc[g][mt][v] = cok ? xgt[m * 2064 + g * 516 + tc] : 0.0f;
          }
      for (int kk = 0; kk < 17; ++kk) {
        const int kb = kk * 32 + hi * 8;
        Frag a0, a1;
        const __bf16* hp = &hbuf[cur][0];
        a0.q[0] = *(const uint4*)(hp + (0  + nrow) * 544 + kb);
        a0.q[1] = *(const uint4*)(hp + (0  + nrow) * 544 + kb + 16);
        a1.q[0] = *(const uint4*)(hp + (16 + nrow) * 544 + kb);
        a1.q[1] = *(const uint4*)(hp + (16 + nrow) * 544 + kb + 16);
        for (int g = 0; g < 4; ++g) {
          Frag b;
          const __bf16* bp = whh + (long)(g * 516 + tile * 16 + nrow) * 544 + kb;
          b.q[0] = *(const uint4*)(bp);
          b.q[1] = *(const uint4*)(bp + 16);
          acc[g][0] = WMMA_BF16(a0.v, b.v, acc[g][0]);
          acc[g][1] = WMMA_BF16(a1.v, b.v, acc[g][1]);
        }
      }
      for (int mt = 0; mt < 2; ++mt)
        for (int v = 0; v < 8; ++v) {
          int m = mt * 16 + v + 8 * hi;
          float iv = sigf(acc[0][mt][v]);
          float fv = sigf(acc[1][mt][v]);
          float gv = tanhf(acc[2][mt][v]);
          float ov = sigf(acc[3][mt][v]);
          float cn = fv * creg[slot][mt][v] + iv * gv;
          creg[slot][mt][v] = cn;
          float hv = ov * tanhf(cn);
          if (cok) {
            __bf16 hb = (__bf16)hv;
            hbuf[nxt][m * 544 + tc] = hb;
            hs[((long)t * 32 + m) * 544 + tc] = hb;
            if (hN != nullptr && t == S - 1) {
              hN[m * 516 + tc] = hv;
              cN[m * 516 + tc] = cn;
            }
          }
        }
    }
    __syncthreads();
  }
}

// ------------------------------- fused projection + log-softmax NLL loss
// A: [4096 x 544] bf16 (dec layer1 h), Wt: [32000 x 544] bf16, outb: [32000].
// Each block handles 128 rows (A tile cached in LDS via async DMA), sweeps
// all 32000 cols in 64-wide chunks with online max/sum;
// loss += (logsumexp - logit[tgt]) / 32 per valid row.
__global__ __launch_bounds__(256) void k_proj_loss(
    const __bf16* __restrict__ A, const __bf16* __restrict__ Wt,
    const float* __restrict__ outb, const int* __restrict__ y,
    float* __restrict__ loss) {
  __shared__ __align__(16) __bf16 atile[128 * 544];
  __shared__ __align__(16) __bf16 btile[2][64 * 32];
  const int tid = threadIdx.x;
  const int w = tid >> 5, lane = tid & 31;
  const int nrow = lane & 15, hi = lane >> 4;
  const int mbase = blockIdx.x * 128;
  const int nl = tid >> 2, kc = (tid & 3) * 8;

  for (int i = tid; i < 128 * 68; i += 256) {
    int r = i / 68, cchunk = (i % 68) * 8;
    async_copy16(A + (long)(mbase + r) * 544 + cchunk, &atile[r * 544 + cchunk]);
  }
  WAIT_ASYNC0();
  __syncthreads();

  const int m0l = w * 16;
  float rm[8], rs[8], tl[8];
  int tg[8];
  for (int v = 0; v < 8; ++v) {
    rm[v] = -3.0e38f; rs[v] = 0.0f; tl[v] = 0.0f;
    int m = mbase + m0l + v + 8 * hi;
    tg[v] = (m < 4064) ? y[m + 32] : -1;
  }

  for (int nc = 0; nc < 500; ++nc) {
    const int n0 = nc * 64;
    const __bf16* brow = Wt + (long)(n0 + nl) * 544 + kc;
    v8f acc[4];
    for (int nt = 0; nt < 4; ++nt)
      for (int v = 0; v < 8; ++v) acc[nt][v] = 0.0f;

    async_copy16(brow, &btile[0][nl * 32 + kc]);   // stage k-slab 0
    for (int kk = 0; kk < 17; ++kk) {
      const int cur = kk & 1;
      if (kk + 1 < 17) {
        async_copy16(brow + (kk + 1) * 32, &btile[cur ^ 1][nl * 32 + kc]);
        WAIT_ASYNC1();
      } else {
        WAIT_ASYNC0();
      }
      __syncthreads();
      Frag a;
      const int kb = kk * 32 + hi * 8;
      const __bf16* ap = atile + (m0l + nrow) * 544 + kb;
      a.q[0] = *(const uint4*)(ap);
      a.q[1] = *(const uint4*)(ap + 16);
      const __bf16* bt = &btile[cur][0];
      for (int nt = 0; nt < 4; ++nt) {
        Frag b;
        const __bf16* bp = bt + (nt * 16 + nrow) * 32 + hi * 8;
        b.q[0] = *(const uint4*)(bp);
        b.q[1] = *(const uint4*)(bp + 16);
        acc[nt] = WMMA_BF16(a.v, b.v, acc[nt]);
      }
      __syncthreads();
    }
    float bb[4];
    for (int nt = 0; nt < 4; ++nt) bb[nt] = outb[n0 + nt * 16 + nrow];
    for (int v = 0; v < 8; ++v) {
      float lv[4];
      float cmax = -3.0e38f;
      for (int nt = 0; nt < 4; ++nt) {
        int n = n0 + nt * 16 + nrow;
        lv[nt] = acc[nt][v] + bb[nt];
        cmax = fmaxf(cmax, lv[nt]);
        if (n == tg[v]) tl[v] = lv[nt];
      }
      for (int mask = 1; mask < 16; mask <<= 1)
        cmax = fmaxf(cmax, __shfl_xor(cmax, mask, 32));
      float nm = fmaxf(rm[v], cmax);
      float es = 0.0f;
      for (int nt = 0; nt < 4; ++nt) es += __expf(lv[nt] - nm);
      for (int mask = 1; mask < 16; mask <<= 1)
        es += __shfl_xor(es, mask, 32);
      rs[v] = rs[v] * __expf(rm[v] - nm) + es;
      rm[v] = nm;
    }
  }
  for (int v = 0; v < 8; ++v) {
    float t = tl[v];
    for (int mask = 1; mask < 16; mask <<= 1) t += __shfl_xor(t, mask, 32);
    int m = mbase + m0l + v + 8 * hi;
    if ((lane & 15) == 0 && m < 4064) {
      float nll = (rm[v] + logf(rs[v])) - t;
      atomicAdd(loss, nll * (1.0f / 32.0f));
    }
  }
}

// ---------------------------------------------------------------- driver
extern "C" void kernel_launch(void* const* d_in, const int* in_sizes, int n_in,
                              void* d_out, int out_size, void* d_ws, size_t ws_size,
                              hipStream_t stream) {
  const int*   x       = (const int*)d_in[0];
  const int*   y       = (const int*)d_in[1];
  const float* enc_emb = (const float*)d_in[2];
  const float* dec_emb = (const float*)d_in[3];
  const float* enc_w0i = (const float*)d_in[4];
  const float* enc_w0h = (const float*)d_in[5];
  const float* enc_b0  = (const float*)d_in[6];
  const float* enc_w1i = (const float*)d_in[7];
  const float* enc_w1h = (const float*)d_in[8];
  const float* enc_b1  = (const float*)d_in[9];
  const float* dec_w0i = (const float*)d_in[10];
  const float* dec_w0h = (const float*)d_in[11];
  const float* dec_b0  = (const float*)d_in[12];
  const float* dec_w1i = (const float*)d_in[13];
  const float* dec_w1h = (const float*)d_in[14];
  const float* dec_b1  = (const float*)d_in[15];
  const float* out_w   = (const float*)d_in[16];
  const float* out_b   = (const float*)d_in[17];
  float* loss = (float*)d_out;

  char* ws = (char*)d_ws;
  size_t off = 0;
  auto take = [&](size_t bytes) -> void* {
    void* p = ws + off;
    off += (bytes + 255) & ~(size_t)255;
    return p;
  };
  // bf16 weights (rows padded to 64, K padded to 32)
  __bf16* wb_e0i = (__bf16*)take((size_t)2112 * 256 * 2);
  __bf16* wb_e0h = (__bf16*)take((size_t)2112 * 544 * 2);
  __bf16* wb_e1i = (__bf16*)take((size_t)2112 * 544 * 2);
  __bf16* wb_e1h = (__bf16*)take((size_t)2112 * 544 * 2);
  __bf16* wb_d0i = (__bf16*)take((size_t)2112 * 256 * 2);
  __bf16* wb_d0h = (__bf16*)take((size_t)2112 * 544 * 2);
  __bf16* wb_d1i = (__bf16*)take((size_t)2112 * 544 * 2);
  __bf16* wb_d1h = (__bf16*)take((size_t)2112 * 544 * 2);
  __bf16* wb_out = (__bf16*)take((size_t)32000 * 544 * 2);
  // bf16 activations
  __bf16* ex   = (__bf16*)take((size_t)4096 * 256 * 2);
  __bf16* ey   = (__bf16*)take((size_t)4096 * 256 * 2);
  __bf16* hs0  = (__bf16*)take((size_t)4096 * 544 * 2);
  __bf16* d0b  = (__bf16*)take((size_t)4096 * 544 * 2);
  __bf16* d1b  = (__bf16*)take((size_t)4096 * 544 * 2);
  // fp32 gate pre-activations (reused per layer) and saved states
  float* xg  = (float*)take((size_t)4096 * 2064 * 4);
  float* he0 = (float*)take((size_t)32 * 516 * 4);
  float* ce0 = (float*)take((size_t)32 * 516 * 4);
  float* he1 = (float*)take((size_t)32 * 516 * 4);
  float* ce1 = (float*)take((size_t)32 * 516 * 4);
  (void)ws_size; (void)in_sizes; (void)n_in; (void)out_size;

  auto g1 = [](long n) { return dim3((unsigned)((n + 255) / 256)); };

  // weight casts
  k_cast_pad<<<g1((long)2112 * 256), 256, 0, stream>>>(enc_w0i, wb_e0i, 2064, 256, 2112, 256);
  k_cast_pad<<<g1((long)2112 * 544), 256, 0, stream>>>(enc_w0h, wb_e0h, 2064, 516, 2112, 544);
  k_cast_pad<<<g1((long)2112 * 544), 256, 0, stream>>>(enc_w1i, wb_e1i, 2064, 516, 2112, 544);
  k_cast_pad<<<g1((long)2112 * 544), 256, 0, stream>>>(enc_w1h, wb_e1h, 2064, 516, 2112, 544);
  k_cast_pad<<<g1((long)2112 * 256), 256, 0, stream>>>(dec_w0i, wb_d0i, 2064, 256, 2112, 256);
  k_cast_pad<<<g1((long)2112 * 544), 256, 0, stream>>>(dec_w0h, wb_d0h, 2064, 516, 2112, 544);
  k_cast_pad<<<g1((long)2112 * 544), 256, 0, stream>>>(dec_w1i, wb_d1i, 2064, 516, 2112, 544);
  k_cast_pad<<<g1((long)2112 * 544), 256, 0, stream>>>(dec_w1h, wb_d1h, 2064, 516, 2112, 544);
  k_cast_pad<<<g1((long)32000 * 544), 256, 0, stream>>>(out_w, wb_out, 32000, 516, 32000, 544);

  // embeddings
  k_gather_cast<<<g1((long)4096 * 256), 256, 0, stream>>>(x, enc_emb, ex, 4096, 4096);
  k_gather_cast<<<g1((long)4096 * 256), 256, 0, stream>>>(y, dec_emb, ey, 4064, 4096);

  // zero h-state buffers (guarantees K-pad columns and M-pad rows are 0)
  k_zero_bf16<<<g1((long)4096 * 544), 256, 0, stream>>>(hs0, 4096 * 544);
  k_zero_bf16<<<g1((long)4096 * 544), 256, 0, stream>>>(d0b, 4096 * 544);
  k_zero_bf16<<<g1((long)4096 * 544), 256, 0, stream>>>(d1b, 4096 * 544);

  const dim3 gblk(33, 32), gthr(256);
  // encoder layer 0
  k_gemm_bias<<<gblk, gthr, 0, stream>>>(ex, wb_e0i, enc_b0, xg, 4096, 2064, 2064, 256);
  k_lstm_rec<<<1, 1024, 0, stream>>>(xg, wb_e0h, nullptr, nullptr, hs0, he0, ce0, 128);
  // encoder layer 1 (hidden sequence discarded; only final h,c used)
  k_gemm_bias<<<gblk, gthr, 0, stream>>>(hs0, wb_e1i, enc_b1, xg, 4096, 2064, 2064, 544);
  k_lstm_rec<<<1, 1024, 0, stream>>>(xg, wb_e1h, nullptr, nullptr, d0b, he1, ce1, 128);
  // decoder layer 0
  k_gemm_bias<<<gblk, gthr, 0, stream>>>(ey, wb_d0i, dec_b0, xg, 4096, 2064, 2064, 256);
  k_lstm_rec<<<1, 1024, 0, stream>>>(xg, wb_d0h, he0, ce0, d0b, nullptr, nullptr, 127);
  // decoder layer 1
  k_gemm_bias<<<gblk, gthr, 0, stream>>>(d0b, wb_d1i, dec_b1, xg, 4096, 2064, 2064, 544);
  k_lstm_rec<<<1, 1024, 0, stream>>>(xg, wb_d1h, he1, ce1, d1b, nullptr, nullptr, 127);

  // fused projection + log-softmax NLL
  k_zero_f32<<<1, 1, 0, stream>>>(loss, 1);
  k_proj_loss<<<32, 256, 0, stream>>>(d1b, wb_out, out_b, y, loss);
}
